// MultiHeadAttentionClassifierPyTorch_76373108457864
// MI455X (gfx1250) — matrix operand order repaired and checked by reading, not compiled, not run on hardware
//
#include <hip/hip_runtime.h>
#include <hip/hip_bf16.h>
#include <math.h>

// ---------------------------------------------------------------------------
// Types for CDNA5 WMMA (wave32). v_wmma_f32_16x16x32_bf16:
//   (bool neg_a, v16bf A, bool neg_b, v16bf B, short c_mod, v8f C, bool, bool)
// ---------------------------------------------------------------------------
typedef __attribute__((ext_vector_type(16))) __bf16 v16bf;
typedef __attribute__((ext_vector_type(8)))  __bf16 v8bf;
typedef __attribute__((ext_vector_type(8)))  float  v8f;

// Problem constants (from reference)
#define DMODEL 2048
#define NHEAD  8
#define HD_ENC 256      // DMODEL / NHEAD
#define KS     512      // final MHA per-head dim
#define HKS    4096     // NHEAD * KS
#define FFH    512
#define FRAMES 5
#define BATCH  4096
#define BFROWS (BATCH * FRAMES)   // 20480 token rows

// GEMM tiling: block tile 128x256, K-step 32, 8 waves (2 M x 4 N), each 64x64
#define BM 128
#define BN 256
#define BK 32
#define LPAD 8
#define LSTR (BK + LPAD)          // LDS row stride in bf16 (80B, 16B aligned)

// ---------------------------------------------------------------------------
// CDNA5 async global->LDS DMA (ASYNCcnt-tracked), per-lane 16B move.
// Generic LDS pointer low 32 bits == LDS byte address (flat aperture rule).
// ---------------------------------------------------------------------------
__device__ __forceinline__ void async_ld16(void* lds_dst, const void* gsrc) {
  unsigned loff = (unsigned)(size_t)lds_dst;
  asm volatile("global_load_async_to_lds_b128 %0, %1, off"
               :: "v"(loff), "v"(gsrc) : "memory");
}
__device__ __forceinline__ void wait_async0() {
  asm volatile("s_wait_asynccnt 0x0" ::: "memory");
}

// ---------------------------------------------------------------------------
// f32 -> bf16 conversion (grid-stride)
// ---------------------------------------------------------------------------
__global__ __launch_bounds__(256)
void cvt_f32_bf16(const float* __restrict__ s, __bf16* __restrict__ d, size_t n) {
  size_t i = (size_t)blockIdx.x * blockDim.x + threadIdx.x;
  size_t stride = (size_t)gridDim.x * blockDim.x;
  for (; i < n; i += stride) d[i] = (__bf16)s[i];
}

// ---------------------------------------------------------------------------
// C[M,N] = act( A[M,K](bf16) @ W[N,K](bf16)^T + bias )  -> f32 and/or bf16
// K % 32 == 0 (true for all calls). OOB tile rows are clamped (dup data is
// never stored: epilogue guards rows/cols), keeping staging branch-free.
// Double-buffered LDS fed by async DMA; compute overlaps the next fetch.
// ---------------------------------------------------------------------------
__global__ __launch_bounds__(256)
void wmma_gemm_bf16(const __bf16* __restrict__ A, const __bf16* __restrict__ W,
                    const float* __restrict__ bias,
                    float* __restrict__ Cf, __bf16* __restrict__ Cb,
                    int M, int N, int K, int relu)
{
  __shared__ __bf16 As[2][BM * LSTR];   // 2 x 10 KB
  __shared__ __bf16 Ws[2][BN * LSTR];   // 2 x 20 KB

  const int tid     = threadIdx.x;
  const int lane    = tid & 31;
  const int wid     = tid >> 5;
  const int lane_lo = lane & 15;
  const int lane_hi = lane >> 4;

  const int wm = (wid & 1) * 64;   // wave M offset within block tile
  const int wn = (wid >> 1) * 64;  // wave N offset within block tile

  const int m0 = blockIdx.y * BM;
  const int n0 = blockIdx.x * BN;

  // Per-thread staging chunk coordinates (8 bf16 = 16B per chunk).
  // A tile: 512 chunks (2/thread). W tile: 1024 chunks (4/thread).
  int a_r[2], a_c[2], w_r[4], w_c[4];
  const __bf16* a_g[2];
  const __bf16* w_g[4];
  #pragma unroll
  for (int c = 0; c < 2; c++) {
    int chunk = tid + c * 256;
    a_r[c] = chunk >> 2; a_c[c] = (chunk & 3) * 8;
    int gr = m0 + a_r[c]; if (gr > M - 1) gr = M - 1;
    a_g[c] = A + (size_t)gr * K + a_c[c];
  }
  #pragma unroll
  for (int c = 0; c < 4; c++) {
    int chunk = tid + c * 256;
    w_r[c] = chunk >> 2; w_c[c] = (chunk & 3) * 8;
    int gn = n0 + w_r[c]; if (gn > N - 1) gn = N - 1;
    w_g[c] = W + (size_t)gn * K + w_c[c];
  }

  v8f acc[4][4] = {};

  // Prologue: DMA tile k0=0 into buffer 0.
  #pragma unroll
  for (int c = 0; c < 2; c++)
    async_ld16(&As[0][a_r[c] * LSTR + a_c[c]], a_g[c]);
  #pragma unroll
  for (int c = 0; c < 4; c++)
    async_ld16(&Ws[0][w_r[c] * LSTR + w_c[c]], w_g[c]);

  int buf = 0;
  for (int k0 = 0; k0 < K; k0 += BK) {
    wait_async0();      // own DMA into As/Ws[buf] has landed
    __syncthreads();    // everyone's DMA landed; prior reads of buf^1 done

    if (k0 + BK < K) {
      int nb = buf ^ 1;
      int koff = k0 + BK;
      #pragma unroll
      for (int c = 0; c < 2; c++)
        async_ld16(&As[nb][a_r[c] * LSTR + a_c[c]], a_g[c] + koff);
      #pragma unroll
      for (int c = 0; c < 4; c++)
        async_ld16(&Ws[nb][w_r[c] * LSTR + w_c[c]], w_g[c] + koff);
    }

    // A fragments: 16x32 bf16. Lanes 0-15 (M=lane): K=0..7 then 16..23;
    // lanes 16-31: K=8..15 then 24..31.
    v16bf afrag[4];
    #pragma unroll
    for (int t = 0; t < 4; t++) {
      const __bf16* ap = &As[buf][(wm + t * 16 + lane_lo) * LSTR + lane_hi * 8];
      v8bf lo = *(const v8bf*)ap;
      v8bf hi = *(const v8bf*)(ap + 16);
      afrag[t] = __builtin_shufflevector(lo, hi,
          0,1,2,3,4,5,6,7,8,9,10,11,12,13,14,15);
    }

    // B fragments: 32x16 (= W^T tile). Lane n (0-15): K=0..7 then 16..23;
    // lanes 16-31: K=8..15 then 24..31.
    #pragma unroll
    for (int j = 0; j < 4; j++) {
      const __bf16* bp = &Ws[buf][(wn + j * 16 + lane_lo) * LSTR + lane_hi * 8];
      v8bf b0 = *(const v8bf*)bp;
      v8bf b1 = *(const v8bf*)(bp + 16);
      v16bf bfrag = __builtin_shufflevector(b0, b1,
          0,1,2,3,4,5,6,7,8,9,10,11,12,13,14,15);
      #pragma unroll
      for (int t = 0; t < 4; t++) {
        acc[t][j] = __builtin_amdgcn_wmma_f32_16x16x32_bf16(
            false, afrag[t], false, bfrag, (short)0, acc[t][j], false, false);
      }
    }
    buf ^= 1;
  }

  // Epilogue. D layout: VGPR r -> M = lane_hi*8 + r, N = lane_lo.
  #pragma unroll
  for (int t = 0; t < 4; t++) {
    int row_base = m0 + wm + t * 16 + lane_hi * 8;
    #pragma unroll
    for (int j = 0; j < 4; j++) {
      int col = n0 + wn + j * 16 + lane_lo;
      if (col >= N) continue;
      float bv = bias ? bias[col] : 0.0f;
      #pragma unroll
      for (int r = 0; r < 8; r++) {
        int row = row_base + r;
        if (row >= M) continue;
        float v = acc[t][j][r] + bv;
        if (relu) v = fmaxf(v, 0.0f);
        size_t idx = (size_t)row * N + col;
        if (Cf) Cf[idx] = v;
        if (Cb) Cb[idx] = (__bf16)v;
      }
    }
  }
}

// ---------------------------------------------------------------------------
// Tiny attention: per (batch, head) block, F<=8 frames, softmax(QK^T*s)V.
// ---------------------------------------------------------------------------
__global__ __launch_bounds__(256)
void sdpa_small(const __bf16* __restrict__ Qb, const __bf16* __restrict__ Kb,
                const __bf16* __restrict__ Vb, __bf16* __restrict__ Ob,
                int Fr, int HD, int H, long rs_in, long rs_out, float scale)
{
  const int b = blockIdx.x / H;
  const int h = blockIdx.x % H;
  const __bf16* q = Qb + (size_t)b * Fr * rs_in + (size_t)h * HD;
  const __bf16* k = Kb + (size_t)b * Fr * rs_in + (size_t)h * HD;
  const __bf16* v = Vb + (size_t)b * Fr * rs_in + (size_t)h * HD;
  __bf16*       o = Ob + (size_t)b * Fr * rs_out + (size_t)h * HD;

  __shared__ float sS[64];
  __shared__ float sP[64];

  const int lane = threadIdx.x & 31;
  const int wid  = threadIdx.x >> 5;

  for (int p = wid; p < Fr * Fr; p += 8) {
    int i = p / Fr, j = p % Fr;
    float s = 0.0f;
    for (int d = lane; d < HD; d += 32)
      s += (float)q[(size_t)i * rs_in + d] * (float)k[(size_t)j * rs_in + d];
    #pragma unroll
    for (int m = 16; m > 0; m >>= 1) s += __shfl_xor(s, m, 32);
    if (lane == 0) sS[p] = s * scale;
  }
  __syncthreads();

  if ((int)threadIdx.x < Fr) {
    int i = threadIdx.x;
    float mx = -3.0e38f;
    for (int j = 0; j < Fr; j++) mx = fmaxf(mx, sS[i * Fr + j]);
    float sum = 0.0f;
    for (int j = 0; j < Fr; j++) { float e = __expf(sS[i * Fr + j] - mx); sP[i * Fr + j] = e; sum += e; }
    float inv = 1.0f / sum;
    for (int j = 0; j < Fr; j++) sP[i * Fr + j] *= inv;
  }
  __syncthreads();

  for (int d = threadIdx.x; d < HD; d += 256) {
    for (int i = 0; i < Fr; i++) {
      float accv = 0.0f;
      for (int j = 0; j < Fr; j++)
        accv += sP[i * Fr + j] * (float)v[(size_t)j * rs_in + d];
      o[(size_t)i * rs_out + d] = (__bf16)accv;
    }
  }
}

// ---------------------------------------------------------------------------
// out = LayerNorm(A + R) * g + b ; writes f32 (optional) and bf16 (optional).
// One 256-thread block per row. Safe when outf aliases R (same-thread RAW).
// ---------------------------------------------------------------------------
__global__ __launch_bounds__(256)
void add_layernorm(const float* __restrict__ A, const float* __restrict__ R,
                   const float* __restrict__ g, const float* __restrict__ bta,
                   float* __restrict__ outf, __bf16* __restrict__ outb, int Dlen)
{
  const size_t row = blockIdx.x;
  const float* a = A + row * (size_t)Dlen;
  const float* r = R + row * (size_t)Dlen;

  float s = 0.0f, s2 = 0.0f;
  for (int i = threadIdx.x; i < Dlen; i += 256) {
    float x = a[i] + r[i];
    s += x; s2 += x * x;
  }
  #pragma unroll
  for (int m = 16; m > 0; m >>= 1) { s += __shfl_xor(s, m, 32); s2 += __shfl_xor(s2, m, 32); }

  __shared__ float red[2][8];
  const int lane = threadIdx.x & 31, wid = threadIdx.x >> 5;
  if (lane == 0) { red[0][wid] = s; red[1][wid] = s2; }
  __syncthreads();
  __shared__ float stats[2];
  if (threadIdx.x == 0) {
    float ts = 0.0f, ts2 = 0.0f;
    for (int w = 0; w < 8; w++) { ts += red[0][w]; ts2 += red[1][w]; }
    float mean = ts / (float)Dlen;
    float var  = ts2 / (float)Dlen - mean * mean;
    stats[0] = mean;
    stats[1] = rsqrtf(var + 1e-5f);
  }
  __syncthreads();
  const float mean = stats[0], inv = stats[1];

  for (int i = threadIdx.x; i < Dlen; i += 256) {
    float x = a[i] + r[i];
    float y = (x - mean) * inv * g[i] + bta[i];
    if (outf) outf[row * (size_t)Dlen + i] = y;
    if (outb) outb[row * (size_t)Dlen + i] = (__bf16)y;
  }
}

// ---------------------------------------------------------------------------
// Host side
// ---------------------------------------------------------------------------
static inline int cvt_blocks(size_t n) {
  size_t b = (n + 2047) / 2048;
  if (b > 65535u * 16u) b = 65535u * 16u;
  return (int)b;
}

extern "C" void kernel_launch(void* const* d_in, const int* in_sizes, int n_in,
                              void* d_out, int out_size, void* d_ws, size_t ws_size,
                              hipStream_t stream) {
  (void)in_sizes; (void)n_in; (void)out_size; (void)ws_size;

  const float* x        = (const float*)d_in[0];
  const float* inproj_w = (const float*)d_in[1];
  const float* inproj_b = (const float*)d_in[2];
  const float* encout_w = (const float*)d_in[3];
  const float* encout_b = (const float*)d_in[4];
  const float* ln1_g    = (const float*)d_in[5];
  const float* ln1_b    = (const float*)d_in[6];
  const float* ff1_w    = (const float*)d_in[7];
  const float* ff1_b    = (const float*)d_in[8];
  const float* ff2_w    = (const float*)d_in[9];
  const float* ff2_b    = (const float*)d_in[10];
  const float* ln2_g    = (const float*)d_in[11];
  const float* ln2_b    = (const float*)d_in[12];
  const float* wq       = (const float*)d_in[13];
  const float* wk       = (const float*)d_in[14];
  const float* wv       = (const float*)d_in[15];
  const float* wo       = (const float*)d_in[16];
  const float* bo       = (const float*)d_in[17];
  const float* lnf_g    = (const float*)d_in[18];
  const float* lnf_b    = (const float*)d_in[19];
  const float* mlp_w[5] = { (const float*)d_in[20], (const float*)d_in[21],
                            (const float*)d_in[22], (const float*)d_in[23],
                            (const float*)d_in[24] };
  const float* mlp_b[5] = { (const float*)d_in[25], (const float*)d_in[26],
                            (const float*)d_in[27], (const float*)d_in[28],
                            (const float*)d_in[29] };

  size_t off = 0;
  auto alloc = [&](size_t bytes) -> void* {
    void* p = (char*)d_ws + off;
    off += (bytes + 255) & ~(size_t)255;
    return p;
  };

  const size_t SZ[6] = { 10240, 512, 256, 128, 64, 8 };

  __bf16* wb_inproj = (__bf16*)alloc((size_t)3 * DMODEL * DMODEL * 2);
  __bf16* wb_encout = (__bf16*)alloc((size_t)DMODEL * DMODEL * 2);
  __bf16* wb_ff1    = (__bf16*)alloc((size_t)FFH * DMODEL * 2);
  __bf16* wb_ff2    = (__bf16*)alloc((size_t)DMODEL * FFH * 2);
  __bf16* wb_q      = (__bf16*)alloc((size_t)HKS * DMODEL * 2);
  __bf16* wb_k      = (__bf16*)alloc((size_t)HKS * DMODEL * 2);
  __bf16* wb_v      = (__bf16*)alloc((size_t)HKS * DMODEL * 2);
  __bf16* wb_o      = (__bf16*)alloc((size_t)DMODEL * HKS * 2);
  __bf16* wb_m[5];
  for (int i = 0; i < 5; i++) wb_m[i] = (__bf16*)alloc(SZ[i] * SZ[i + 1] * 2);

  __bf16* xb     = (__bf16*)alloc((size_t)BFROWS * DMODEL * 2);
  __bf16* qkvb   = (__bf16*)alloc((size_t)BFROWS * 3 * DMODEL * 2);
  __bf16* attnb  = (__bf16*)alloc((size_t)BFROWS * DMODEL * 2);
  float*  gscr   = (float*) alloc((size_t)BFROWS * DMODEL * 4);  // reused 3x
  float*  h1f    = (float*) alloc((size_t)BFROWS * DMODEL * 4);  // h2f aliases
  __bf16* h1b    = (__bf16*)alloc((size_t)BFROWS * DMODEL * 2);  // h2b aliases
  __bf16* ff1b   = (__bf16*)alloc((size_t)BFROWS * FFH * 2);
  __bf16* q2b    = (__bf16*)alloc((size_t)BFROWS * HKS * 2);
  __bf16* k2b    = (__bf16*)alloc((size_t)BFROWS * HKS * 2);
  __bf16* v2b    = (__bf16*)alloc((size_t)BFROWS * HKS * 2);
  __bf16* o2b    = (__bf16*)alloc((size_t)BFROWS * HKS * 2);
  __bf16* h3b    = attnb;                                        // reuse
  __bf16* z1     = (__bf16*)alloc((size_t)BATCH * 512 * 2);
  __bf16* z2     = (__bf16*)alloc((size_t)BATCH * 256 * 2);
  __bf16* z3     = (__bf16*)alloc((size_t)BATCH * 128 * 2);
  __bf16* z4     = (__bf16*)alloc((size_t)BATCH * 64 * 2);
  float*  h2f = h1f;
  __bf16* h2b = h1b;

  auto cvt = [&](const float* s, __bf16* d, size_t n) {
    cvt_f32_bf16<<<cvt_blocks(n), 256, 0, stream>>>(s, d, n);
  };
  auto gemm = [&](const __bf16* A, const __bf16* Wt, const float* bias,
                  float* Cf, __bf16* Cb, int M, int N, int K, int relu) {
    dim3 grid((N + BN - 1) / BN, (M + BM - 1) / BM);
    wmma_gemm_bf16<<<grid, 256, 0, stream>>>(A, Wt, bias, Cf, Cb, M, N, K, relu);
  };

  cvt(x, xb, (size_t)BFROWS * DMODEL);
  cvt(inproj_w, wb_inproj, (size_t)3 * DMODEL * DMODEL);
  cvt(encout_w, wb_encout, (size_t)DMODEL * DMODEL);
  cvt(ff1_w, wb_ff1, (size_t)FFH * DMODEL);
  cvt(ff2_w, wb_ff2, (size_t)DMODEL * FFH);
  cvt(wq, wb_q, (size_t)HKS * DMODEL);
  cvt(wk, wb_k, (size_t)HKS * DMODEL);
  cvt(wv, wb_v, (size_t)HKS * DMODEL);
  cvt(wo, wb_o, (size_t)DMODEL * HKS);
  for (int i = 0; i < 5; i++) cvt(mlp_w[i], wb_m[i], SZ[i] * SZ[i + 1]);

  // ---- encoder layer ----
  gemm(xb, wb_inproj, inproj_b, nullptr, qkvb, BFROWS, 3 * DMODEL, DMODEL, 0);
  sdpa_small<<<BATCH * NHEAD, 256, 0, stream>>>(
      qkvb, qkvb + DMODEL, qkvb + 2 * DMODEL, attnb,
      FRAMES, HD_ENC, NHEAD, 3 * DMODEL, DMODEL, 1.0f / 16.0f);
  gemm(attnb, wb_encout, encout_b, gscr, nullptr, BFROWS, DMODEL, DMODEL, 0);
  add_layernorm<<<BFROWS, 256, 0, stream>>>(gscr, x, ln1_g, ln1_b, h1f, h1b, DMODEL);

  gemm(h1b, wb_ff1, ff1_b, nullptr, ff1b, BFROWS, FFH, DMODEL, 1);
  gemm(ff1b, wb_ff2, ff2_b, gscr, nullptr, BFROWS, DMODEL, FFH, 0);
  add_layernorm<<<BFROWS, 256, 0, stream>>>(gscr, h1f, ln2_g, ln2_b, h2f, h2b, DMODEL);

  // ---- final MHA ----
  gemm(h2b, wb_q, nullptr, nullptr, q2b, BFROWS, HKS, DMODEL, 0);
  gemm(h2b, wb_k, nullptr, nullptr, k2b, BFROWS, HKS, DMODEL, 0);
  gemm(h2b, wb_v, nullptr, nullptr, v2b, BFROWS, HKS, DMODEL, 0);
  sdpa_small<<<BATCH * NHEAD, 256, 0, stream>>>(
      q2b, k2b, v2b, o2b, FRAMES, KS, NHEAD, HKS, HKS, 0.044194173824159216f);
  gemm(o2b, wb_o, bo, gscr, nullptr, BFROWS, DMODEL, HKS, 0);
  add_layernorm<<<BFROWS, 256, 0, stream>>>(gscr, h2f, lnf_g, lnf_b, nullptr, h3b, DMODEL);

  // ---- MLP head (h3b is [BATCH, FRAMES*DMODEL] contiguous) ----
  gemm(h3b, wb_m[0], mlp_b[0], nullptr, z1, BATCH, 512, 10240, 1);
  gemm(z1, wb_m[1], mlp_b[1], nullptr, z2, BATCH, 256, 512, 1);
  gemm(z2, wb_m[2], mlp_b[2], nullptr, z3, BATCH, 128, 256, 1);
  gemm(z3, wb_m[3], mlp_b[3], nullptr, z4, BATCH, 64, 128, 1);
  gemm(z4, wb_m[4], mlp_b[4], (float*)d_out, nullptr, BATCH, 8, 64, 0);
}
